// MultiblockConv_74526272520585
// MI455X (gfx1250) — matrix operand order repaired and checked by reading, not compiled, not run on hardware
//
#include <hip/hip_runtime.h>

typedef __bf16 bf16_t;
typedef __bf16 v16bf __attribute__((ext_vector_type(16)));
typedef __bf16 v8bf  __attribute__((ext_vector_type(8)));
typedef float  v8f   __attribute__((ext_vector_type(8)));

#define CIN   64
#define COUT  128
#define HH    256
#define WW    256
#define KTOT  (CIN * 9)     /* 576 */
#define KC    32            /* K chunk staged in LDS (one bf16 WMMA K-step) */
#define NCHUNK (KTOT / KC)  /* 18 */
#define NTILE 64            /* output columns per workgroup */
#define AKS   40            /* LDS row stride in bf16: 80B -> 16B aligned, good banks */
#define NWELT (COUT * KTOT) /* 73728 weight elements */

// ---------------- weight pre-split: fp32 -> bf16 hi + bf16 lo ----------------
__global__ __launch_bounds__(256)
void prepack_w(const float* __restrict__ w, bf16_t* __restrict__ hi, bf16_t* __restrict__ lo)
{
    const int i = blockIdx.x * 256 + threadIdx.x;
    if (i < NWELT) {
        const float f = w[i];
        const bf16_t h = (bf16_t)f;
        hi[i] = h;
        lo[i] = (bf16_t)(f - (float)h);
    }
}

// A 16x32 bf16 fragment: lane half h -> K = 16h..16h+15 (contiguous)
__device__ __forceinline__ v16bf load_fragA(const bf16_t* row, int h)
{
    union { v16bf v; v8bf p[2]; } u;
    u.p[0] = *(const v8bf*)(row + 16 * h);
    u.p[1] = *(const v8bf*)(row + 16 * h + 8);
    return u.v;
}
// B 32x16 bf16 fragment: lane half h -> K = 16h..16h+15 (contiguous)
__device__ __forceinline__ v16bf load_fragB(const bf16_t* row, int h)
{
    union { v16bf v; v8bf p[2]; } u;
    u.p[0] = *(const v8bf*)(row + 16 * h);
    u.p[1] = *(const v8bf*)(row + 16 * h + 8);
    return u.v;
}

#define WMMA_BF16(ACC, A, B) \
    ACC = __builtin_amdgcn_wmma_f32_16x16x32_bf16(false, (A), false, (B), (short)0, (ACC), false, false)

// Implicit-GEMM 3x3 multiblock conv, bf16x2-split (hi/lo) with fp32 accumulate,
// double-buffered LDS + async-to-LDS weight copy + register-staged im2col.
__global__ __launch_bounds__(256)
void conv3x3_wmma_bf16x2(const float* __restrict__ x0,
                         const float* __restrict__ x1,
                         const bf16_t* __restrict__ whi,
                         const bf16_t* __restrict__ wlo,
                         const float* __restrict__ bias,
                         float* __restrict__ out)
{
    __shared__ __align__(16) bf16_t AsH[2][COUT][AKS];   // 2 x 10 KB
    __shared__ __align__(16) bf16_t AsL[2][COUT][AKS];   // 2 x 10 KB
    __shared__ __align__(16) bf16_t BsH[2][NTILE][AKS];  // 2 x 5 KB
    __shared__ __align__(16) bf16_t BsL[2][NTILE][AKS];  // 2 x 5 KB   (60 KB total)

    const int tid  = threadIdx.x;
    const int wave = tid >> 5;
    const int lane = tid & 31;
    const int half = lane >> 4;
    const int ll   = lane & 15;

    const int bid = blockIdx.x;
    const int owt = bid & 3;
    const int oh  = (bid >> 2) & 255;
    const int nb  = (bid >> 10) & 1;   // batch index within block
    const int blk = (bid >> 11) & 1;   // which block (x0 or x1)
    const int ow0 = owt * NTILE;

    const float* xs = blk ? x1 : x0;   // self block
    const float* xo = blk ? x0 : x1;   // other block (horizontal halo source)
    const size_t nbase = (size_t)nb * CIN * HH * WW;

    const int mw = wave & 3;
    const int nw = wave >> 2;
    const int m0 = mw * 32;
    const int n0 = nw * 32;

    v8f acc00 = {}, acc01 = {}, acc10 = {}, acc11 = {};

    // B staging coords: thread covers one column n, 8 consecutive k
    const int bn  = tid & 63;
    const int bks = (tid >> 6) * 8;    // 0,8,16,24

    // ---- pipeline helpers (lambdas keep per-chunk state in registers) ----
    auto issue_A = [&](int kc, int p) {
        // async byte-copy of pre-split weight chunk into LDS buffer p
        #pragma unroll
        for (int rep = 0; rep < 4; ++rep) {
            const int u  = rep * 256 + tid;
            const int ar = u >> 9;            // 0 = hi plane, 1 = lo plane
            const int m  = (u >> 2) & 127;
            const int s  = u & 3;
            const bf16_t* g = (ar ? wlo : whi) + (size_t)m * KTOT + kc + s * 8;
            bf16_t* l = (ar ? &AsL[p][0][0] : &AsH[p][0][0]) + m * AKS + s * 8;
            const unsigned loff = (unsigned)(size_t)l;   // low 32 bits = LDS offset
            asm volatile("global_load_async_to_lds_b128 %0, %1, off"
                         :: "v"(loff), "v"(g) : "memory");
        }
    };
    auto load_B = [&](float* vv, int kc) {
        // im2col gather with multiblock padding into registers (no wait here)
        #pragma unroll
        for (int j = 0; j < 8; ++j) {
            const int k  = kc + bks + j;
            const int c  = k / 9;
            const int r9 = k - c * 9;
            const int kh = r9 / 3;
            const int kw = r9 - kh * 3;
            const int r  = oh + kh - 1;            // vertical: zero pad
            const int cc = ow0 + bn + kw - 1;      // horizontal: cross-block halo
            float v = 0.0f;
            if (r >= 0 && r < HH) {
                const float* rowS = xs + nbase + ((size_t)c * HH + r) * WW;
                const float* rowO = xo + nbase + ((size_t)c * HH + r) * WW;
                v = (cc >= 0 && cc < WW) ? rowS[cc]
                    : (cc < 0 ? rowO[WW - 1] : rowO[0]);
            }
            vv[j] = v;
        }
    };

    // ---- prologue: chunk 0 in flight ----
    float breg[8];
    issue_A(0, 0);
    load_B(breg, 0);

    int p = 0;
    for (int i = 0; i < NCHUNK; ++i, p ^= 1) {
        const int kc = i * KC;

        asm volatile("s_wait_asynccnt 0x0" ::: "memory");  // A(i) landed
        __syncthreads();  // all waves see A(i); buffer 1-p free for rewrite

        // store B(i): hi/lo split into LDS buffer p
        {
            union { v8bf v; } uh, ul;
            #pragma unroll
            for (int j = 0; j < 8; ++j) {
                const bf16_t h = (bf16_t)breg[j];
                uh.v[j] = h;
                ul.v[j] = (bf16_t)(breg[j] - (float)h);
            }
            *(v8bf*)&BsH[p][bn][bks] = uh.v;
            *(v8bf*)&BsL[p][bn][bks] = ul.v;
        }

        // prefetch chunk i+1 under this chunk's compute
        if (i + 1 < NCHUNK) {
            issue_A(kc + KC, p ^ 1);
            load_B(breg, kc + KC);
        }

        __syncthreads();  // B(i) stores visible

        // ---- compute chunk i: 4 tiles x 3 split-terms = 12 bf16 WMMAs ----
        const v16bf bH0 = load_fragB(&BsH[p][n0 + ll][0], half);
        const v16bf bL0 = load_fragB(&BsL[p][n0 + ll][0], half);
        const v16bf bH1 = load_fragB(&BsH[p][n0 + 16 + ll][0], half);
        const v16bf bL1 = load_fragB(&BsL[p][n0 + 16 + ll][0], half);

        {
            const v16bf aH0 = load_fragA(&AsH[p][m0 + ll][0], half);
            const v16bf aL0 = load_fragA(&AsL[p][m0 + ll][0], half);
            WMMA_BF16(acc00, aH0, bH0); WMMA_BF16(acc00, aH0, bL0); WMMA_BF16(acc00, aL0, bH0);
            WMMA_BF16(acc01, aH0, bH1); WMMA_BF16(acc01, aH0, bL1); WMMA_BF16(acc01, aL0, bH1);
        }
        {
            const v16bf aH1 = load_fragA(&AsH[p][m0 + 16 + ll][0], half);
            const v16bf aL1 = load_fragA(&AsL[p][m0 + 16 + ll][0], half);
            WMMA_BF16(acc10, aH1, bH0); WMMA_BF16(acc10, aH1, bL0); WMMA_BF16(acc10, aL1, bH0);
            WMMA_BF16(acc11, aH1, bH1); WMMA_BF16(acc11, aH1, bL1); WMMA_BF16(acc11, aL1, bH1);
        }
    }

    // ---- epilogue: bias + store. C/D layout: VGPR v -> M = v + 8*half, N = ll
    const size_t obase = (size_t)((blk * 2 + nb) * COUT) * (HH * WW)
                       + (size_t)oh * WW + ow0;
    v8f accs[2][2] = {{acc00, acc01}, {acc10, acc11}};
    #pragma unroll
    for (int mt = 0; mt < 2; ++mt) {
        #pragma unroll
        for (int nt = 0; nt < 2; ++nt) {
            const int nc = n0 + nt * 16 + ll;
            #pragma unroll
            for (int v = 0; v < 8; ++v) {
                const int m = m0 + mt * 16 + half * 8 + v;
                out[obase + (size_t)m * (HH * WW) + nc] = accs[mt][nt][v] + bias[m];
            }
        }
    }
}

extern "C" void kernel_launch(void* const* d_in, const int* in_sizes, int n_in,
                              void* d_out, int out_size, void* d_ws, size_t ws_size,
                              hipStream_t stream) {
    (void)in_sizes; (void)n_in; (void)out_size; (void)ws_size;
    const float* x0   = (const float*)d_in[0];
    const float* x1   = (const float*)d_in[1];
    const float* wgt  = (const float*)d_in[2];
    const float* bias = (const float*)d_in[3];
    float* out = (float*)d_out;

    bf16_t* whi = (bf16_t*)d_ws;           // 147456 B
    bf16_t* wlo = whi + NWELT;             // 147456 B  (ws >= 288 KiB)

    prepack_w<<<dim3((NWELT + 255) / 256), dim3(256), 0, stream>>>(wgt, whi, wlo);
    // 2 blocks x 2 batch x 256 rows x 4 ow-tiles = 4096 workgroups
    conv3x3_wmma_bf16x2<<<dim3(4096), dim3(256), 0, stream>>>(x0, x1, whi, wlo, bias, out);
}
